// Unpooling_58119497450175
// MI455X (gfx1250) — matrix operand order repaired and checked by reading, not compiled, not run on hardware
//
#include <hip/hip_runtime.h>
#include <hip/hip_bf16.h>
#include <limits.h>

// ---------------------------------------------------------------------------
// GNN coarsening pipeline for MI455X (gfx1250).
// GEMMs (x@W_conv gathered per edge, x@W_self, h@Wg) run on
// v_wmma_f32_16x16x32_bf16 (bf16 in, f32 accum). Scatters use global f32
// atomics into L2-resident accumulators. h[bloom_node]@Wg is factored to
// h@Wg (computed once per node, gathered 4x).
// ---------------------------------------------------------------------------

typedef __attribute__((ext_vector_type(16))) __bf16 v16bf;
typedef __attribute__((ext_vector_type(8)))  __bf16 v8bf;
typedef __attribute__((ext_vector_type(8)))  float  v8f;

static const int N   = 100000;
static const int E   = 1600000;
static const int D   = 128;
static const int K   = 4;
static const int C   = 200000;
static const int E2  = 3200000;
static const int SPH = 16;
static const int NK  = N * K;
static const int OC  = SPH + D;   // 144 columns of the conv output
#define MIN_RADIUS 0.1f

__device__ __forceinline__ v8f wmma_bf16(v16bf a, v16bf b, v8f c) {
  // D = A(16x32 bf16) * B(32x16 bf16) + C(16x16 f32)
  return __builtin_amdgcn_wmma_f32_16x16x32_bf16(false, a, false, b,
                                                 (short)0, c, false, false);
}

// ---------------------------------------------------------------------------
// Init: zero x_new (in d_out), csum, cnt; batch-max := INT_MIN (segment_max
// identity). Must run every launch since we accumulate with atomics.
// ---------------------------------------------------------------------------
__global__ void k_init(float* __restrict__ xnew, float* __restrict__ csum,
                       float* __restrict__ cnt, int* __restrict__ bmax) {
  long long i = (long long)blockIdx.x * blockDim.x + threadIdx.x;
  const long long n_x = (long long)C * D;        // 25.6M
  const long long n_c = (long long)C * 3;
  if (i < n_x)                       { xnew[i] = 0.f; return; }
  i -= n_x;
  if (i < n_c)                       { csum[i] = 0.f; return; }
  i -= n_c;
  if (i < C)                         { cnt[i]  = 0.f; return; }
  i -= C;
  if (i < C)                         { bmax[i] = INT_MIN; }
}

// Cast a contiguous f32 array to bf16.
__global__ void k_cast(const float* __restrict__ src, __bf16* __restrict__ dst,
                       long long n) {
  long long i = (long long)blockIdx.x * blockDim.x + threadIdx.x;
  if (i < n) dst[i] = (__bf16)src[i];
}

// h = out[:, SPH:SPH+D] cast to bf16, compacted to row stride D.
__global__ void k_cast_h(const float* __restrict__ out, __bf16* __restrict__ h) {
  long long i = (long long)blockIdx.x * blockDim.x + threadIdx.x;
  if (i >= (long long)N * D) return;
  int n = (int)(i >> 7), c = (int)(i & 127);
  h[i] = (__bf16)out[(long long)n * OC + SPH + c];
}

// ---------------------------------------------------------------------------
// Pack a row-major f32 weight matrix W[Kdim x NC] into the per-lane bf16
// B-fragment layout for v_wmma_f32_16x16x32_bf16:
//   fragment (nt, kc): lane<16 -> col nt*16+lane,    elems t = K kc*32+(t&7)+(t>>3)*16
//                      lane>=16 -> col nt*16+lane-16, elems t = K kc*32+8+(t&7)+(t>>3)*16
// so each lane reads 16 contiguous bf16 (32B) per fragment at runtime.
// ---------------------------------------------------------------------------
__global__ void k_pack_b(const float* __restrict__ W, __bf16* __restrict__ P,
                         int Kdim, int NC) {
  int total = (NC / 16) * (Kdim / 32) * 32 * 16;
  int idx = blockIdx.x * blockDim.x + threadIdx.x;
  if (idx >= total) return;
  int t     = idx & 15;
  int lane  = (idx >> 4) & 31;
  int rest  = idx >> 9;
  int KC    = Kdim / 32;
  int kc    = rest % KC;
  int nt    = rest / KC;
  int col   = nt * 16 + (lane & 15);
  int k     = kc * 32 + (lane >> 4) * 8 + (t & 7) + (t >> 3) * 16;
  P[idx] = (__bf16)W[(long long)k * NC + col];
}

// ---------------------------------------------------------------------------
// Dense row GEMM: Dst[16*nTiles x NT*16] = A[.. x 128] (bf16) @ Bpacked.
// One wave per 16-row tile; 4 K-steps of 32; NT column tiles.
// ---------------------------------------------------------------------------
template <int NT>
__global__ void k_row_gemm(const __bf16* __restrict__ A,
                           const __bf16* __restrict__ BP,
                           float* __restrict__ Dst, int nTiles) {
  int wave = blockIdx.x * (blockDim.x >> 5) + (threadIdx.x >> 5);
  int lane = threadIdx.x & 31;
  if (wave >= nTiles) return;                 // wave-uniform: EXEC stays all-1s
  const int m = lane & 15, half = lane >> 4;
  const __bf16* arow = A + (long long)(wave * 16 + m) * 128 + half * 8;
  v8f acc[NT] = {};
#pragma unroll
  for (int kc = 0; kc < 4; ++kc) {
    v8bf lo = *(const v8bf*)(arow + kc * 32);        // K = kc*32+half*8 + 0..7
    v8bf hi = *(const v8bf*)(arow + kc * 32 + 16);   // K = .. +16 .. +23
    v16bf a;
#pragma unroll
    for (int t = 0; t < 8; ++t) { a[t] = lo[t]; a[t + 8] = hi[t]; }
#pragma unroll
    for (int nt = 0; nt < NT; ++nt) {
      v16bf b = *(const v16bf*)(BP + ((long long)(nt * 4 + kc) * 32 + lane) * 16);
      acc[nt] = wmma_bf16(a, b, acc[nt]);
    }
  }
  // C/D layout: lane<16: vgpr j -> row j, col lane; lane>=16: row 8+j.
  const int col = lane & 15, rowoff = half * 8, ldd = NT * 16;
#pragma unroll
  for (int nt = 0; nt < NT; ++nt)
#pragma unroll
    for (int j = 0; j < 8; ++j)
      Dst[(long long)(wave * 16 + rowoff + j) * ldd + nt * 16 + col] = acc[nt][j];
}

// ---------------------------------------------------------------------------
// Edge messages: one wave per 16 edges.
//   msg = x_bf[src] @ W_conv  (WMMA, 16x128 @ 128x144)
//   msg *= edge_attr @ V_conv (per-element coefficient)
//   atomicAdd into out[dst].
// ---------------------------------------------------------------------------
__global__ void k_edge_msg(const __bf16* __restrict__ x_bf,
                           const __bf16* __restrict__ WcP,
                           const float* __restrict__ edge_attr,
                           const float* __restrict__ V_conv,
                           const int* __restrict__ src,
                           const int* __restrict__ dst,
                           float* __restrict__ out) {
  int wave = blockIdx.x * (blockDim.x >> 5) + (threadIdx.x >> 5);
  int lane = threadIdx.x & 31;
  if (wave >= E / 16) return;
  const int ebase = wave * 16;
  const int m = lane & 15, half = lane >> 4;
  const int arowid = src[ebase + m];
  const __bf16* arow = x_bf + (long long)arowid * 128 + half * 8;

  v8f acc[9] = {};
#pragma unroll
  for (int kc = 0; kc < 4; ++kc) {
    v8bf lo = *(const v8bf*)(arow + kc * 32);
    v8bf hi = *(const v8bf*)(arow + kc * 32 + 16);
    v16bf a;
#pragma unroll
    for (int t = 0; t < 8; ++t) { a[t] = lo[t]; a[t + 8] = hi[t]; }
#pragma unroll
    for (int nt = 0; nt < 9; ++nt) {
      v16bf b = *(const v16bf*)(WcP + ((long long)(nt * 4 + kc) * 32 + lane) * 16);
      acc[nt] = wmma_bf16(a, b, acc[nt]);
    }
  }

  const int col = lane & 15, rowoff = half * 8;
  float ea0[8], ea1[8], ea2[8];
  int dd[8];
#pragma unroll
  for (int j = 0; j < 8; ++j) {
    int e = ebase + rowoff + j;
    ea0[j] = edge_attr[3LL * e + 0];
    ea1[j] = edge_attr[3LL * e + 1];
    ea2[j] = edge_attr[3LL * e + 2];
    dd[j]  = dst[e];
  }
#pragma unroll
  for (int nt = 0; nt < 9; ++nt) {
    int cl = nt * 16 + col;
    float v0 = V_conv[cl], v1 = V_conv[OC + cl], v2 = V_conv[2 * OC + cl];
#pragma unroll
    for (int j = 0; j < 8; ++j) {
      float coeff = ea0[j] * v0 + ea1[j] * v1 + ea2[j] * v2;
      atomicAdd(out + (long long)dd[j] * OC + cl, acc[nt][j] * coeff);
    }
  }
}

// ---------------------------------------------------------------------------
// Bloom: offsets = (sph @ W_bloom)*0.1; scatter bloom positions / counts /
// batch-max per cluster. One thread per node.
// ---------------------------------------------------------------------------
__global__ void k_bloom(const float* __restrict__ out, const float* __restrict__ pos,
                        const float* __restrict__ W_bloom,
                        const int* __restrict__ cluster_id,
                        const int* __restrict__ batch,
                        float* __restrict__ csum, float* __restrict__ cnt,
                        int* __restrict__ bmax) {
  __shared__ float wb[SPH * K * 3];               // 16x12
  for (int i = threadIdx.x; i < SPH * K * 3; i += blockDim.x) wb[i] = W_bloom[i];
  __syncthreads();
  int n = blockIdx.x * blockDim.x + threadIdx.x;
  if (n >= N) return;
  float sph[SPH];
#pragma unroll
  for (int t = 0; t < SPH; ++t) sph[t] = out[(long long)n * OC + t];
  float px = pos[3LL * n], py = pos[3LL * n + 1], pz = pos[3LL * n + 2];
  int bt = batch[n];
#pragma unroll
  for (int k = 0; k < K; ++k) {
    float ox = 0.f, oy = 0.f, oz = 0.f;
#pragma unroll
    for (int t = 0; t < SPH; ++t) {
      float s = sph[t];
      ox += s * wb[t * 12 + k * 3 + 0];
      oy += s * wb[t * 12 + k * 3 + 1];
      oz += s * wb[t * 12 + k * 3 + 2];
    }
    int cid = cluster_id[(long long)n * K + k];
    atomicAdd(&csum[3LL * cid + 0], px + MIN_RADIUS * ox);
    atomicAdd(&csum[3LL * cid + 1], py + MIN_RADIUS * oy);
    atomicAdd(&csum[3LL * cid + 2], pz + MIN_RADIUS * oz);
    atomicAdd(&cnt[cid], 1.f);
    atomicMax(&bmax[cid], bt);
  }
}

// new_pos = csum / max(cnt,1); new_batch = (float)bmax.
__global__ void k_newpos(const float* __restrict__ csum, const float* __restrict__ cnt,
                         const int* __restrict__ bmax,
                         float* __restrict__ np, float* __restrict__ nb) {
  int c = blockIdx.x * blockDim.x + threadIdx.x;
  if (c >= C) return;
  float inv = 1.f / fmaxf(cnt[c], 1.f);
  np[3LL * c + 0] = csum[3LL * c + 0] * inv;
  np[3LL * c + 1] = csum[3LL * c + 1] * inv;
  np[3LL * c + 2] = csum[3LL * c + 2] * inv;
  nb[c] = (float)bmax[c];
}

// ---------------------------------------------------------------------------
// Gather scatter: one wave per bloom element i (NK total):
//   gea = pos[i/K] - new_pos[cid];  x_new[cid] += hw[i/K] * (gea @ Vg)
// ---------------------------------------------------------------------------
__global__ void k_gather(const float* __restrict__ hw, const float* __restrict__ pos,
                         const float* __restrict__ np, const float* __restrict__ Vg,
                         const int* __restrict__ cluster_id,
                         float* __restrict__ xnew) {
  int wave = blockIdx.x * (blockDim.x >> 5) + (threadIdx.x >> 5);
  int lane = threadIdx.x & 31;
  if (wave >= NK) return;
  int n   = wave >> 2;
  int cid = cluster_id[wave];
  float g0 = pos[3LL * n + 0] - np[3LL * cid + 0];
  float g1 = pos[3LL * n + 1] - np[3LL * cid + 1];
  float g2 = pos[3LL * n + 2] - np[3LL * cid + 2];
#pragma unroll
  for (int q = 0; q < 4; ++q) {
    int col = lane + 32 * q;
    float coeff = g0 * Vg[col] + g1 * Vg[D + col] + g2 * Vg[2 * D + col];
    atomicAdd(&xnew[(long long)cid * D + col], hw[(long long)n * D + col] * coeff);
  }
}

// new_edge_attr[e] = new_pos[dst] - new_pos[src]; one thread per (e, axis).
__global__ void k_nea(const float* __restrict__ np, const int* __restrict__ nei,
                      float* __restrict__ nea) {
  long long i = (long long)blockIdx.x * blockDim.x + threadIdx.x;
  if (i >= 3LL * E2) return;
  int e = (int)(i / 3), t = (int)(i % 3);
  int a = nei[e];          // new_edge_index[0]
  int b = nei[(long long)E2 + e];  // new_edge_index[1]
  nea[i] = np[3LL * b + t] - np[3LL * a + t];
}

// ---------------------------------------------------------------------------
extern "C" void kernel_launch(void* const* d_in, const int* in_sizes, int n_in,
                              void* d_out, int out_size, void* d_ws, size_t ws_size,
                              hipStream_t stream) {
  const float* x         = (const float*)d_in[0];
  const float* pos       = (const float*)d_in[1];
  const float* edge_attr = (const float*)d_in[2];
  const float* W_conv    = (const float*)d_in[3];   // 128 x 144
  const float* V_conv    = (const float*)d_in[4];   // 3 x 144
  const float* W_self    = (const float*)d_in[5];   // 128 x 144
  const float* W_bloom   = (const float*)d_in[6];   // 16 x 12
  const float* Wg        = (const float*)d_in[7];   // 128 x 128
  const float* Vg        = (const float*)d_in[8];   // 3 x 128
  const int*   eidx      = (const int*)d_in[9];     // 2 x E
  const int*   batch     = (const int*)d_in[10];
  const int*   clid      = (const int*)d_in[11];    // NK
  const int*   neidx     = (const int*)d_in[12];    // 2 x E2
  const int* src = eidx, *dst = eidx + E;

  // Output layout (f32, concatenated in return order).
  float* xnew = (float*)d_out;
  float* np   = xnew + (long long)C * D;
  float* nea  = np + 3LL * C;
  float* nb   = nea + 3LL * E2;

  // Workspace carve (aligned 256B).
  char* ws = (char*)d_ws;
  size_t off = 0;
  auto carve = [&](size_t bytes) {
    void* p = ws + off;
    off = (off + bytes + 255) & ~(size_t)255;
    return p;
  };
  __bf16* x_bf = (__bf16*)carve((size_t)N * D * 2);
  __bf16* h_bf = (__bf16*)carve((size_t)N * D * 2);
  float*  out  = (float*)carve((size_t)N * OC * 4);
  float*  hw   = (float*)carve((size_t)N * D * 4);
  float*  csum = (float*)carve((size_t)C * 3 * 4);
  float*  cnt  = (float*)carve((size_t)C * 4);
  int*    bmax = (int*)carve((size_t)C * 4);
  __bf16* WcP  = (__bf16*)carve((size_t)9 * 4 * 512 * 2);
  __bf16* WsP  = (__bf16*)carve((size_t)9 * 4 * 512 * 2);
  __bf16* WgP  = (__bf16*)carve((size_t)8 * 4 * 512 * 2);
  (void)ws_size; (void)in_sizes; (void)n_in; (void)out_size;

  const int TB = 256;
  // 1) init accumulators
  {
    long long tot = (long long)C * D + 3LL * C + C + C;
    k_init<<<(unsigned)((tot + TB - 1) / TB), TB, 0, stream>>>(xnew, csum, cnt, bmax);
  }
  // 2) cast x; pack weights
  k_cast<<<(unsigned)(((long long)N * D + TB - 1) / TB), TB, 0, stream>>>(x, x_bf, (long long)N * D);
  k_pack_b<<<(9 * 4 * 512 + TB - 1) / TB, TB, 0, stream>>>(W_conv, WcP, D, OC);
  k_pack_b<<<(9 * 4 * 512 + TB - 1) / TB, TB, 0, stream>>>(W_self, WsP, D, OC);
  k_pack_b<<<(8 * 4 * 512 + TB - 1) / TB, TB, 0, stream>>>(Wg, WgP, D, D);
  // 3) out = x @ W_self  (writes all of out -> no zero-init needed)
  k_row_gemm<9><<<(N / 16 + 7) / 8, TB, 0, stream>>>(x_bf, WsP, out, N / 16);
  // 4) out += scatter(edge messages)
  k_edge_msg<<<(E / 16) / 8, TB, 0, stream>>>(x_bf, WcP, edge_attr, V_conv, src, dst, out);
  // 5) h (bf16) from out[:, SPH:]
  k_cast_h<<<(unsigned)(((long long)N * D + TB - 1) / TB), TB, 0, stream>>>(out, h_bf);
  // 6) bloom scatter (csum / cnt / batch max)
  k_bloom<<<(N + TB - 1) / TB, TB, 0, stream>>>(out, pos, W_bloom, clid, batch, csum, cnt, bmax);
  // 7) new_pos, new_batch
  k_newpos<<<(C + TB - 1) / TB, TB, 0, stream>>>(csum, cnt, bmax, np, nb);
  // 8) hw = h @ Wg
  k_row_gemm<8><<<(N / 16 + 7) / 8, TB, 0, stream>>>(h_bf, WgP, hw, N / 16);
  // 9) x_new scatter
  k_gather<<<NK / 8, TB, 0, stream>>>(hw, pos, np, Vg, clid, xnew);
  // 10) new_edge_attr
  k_nea<<<(unsigned)((3LL * E2 + TB - 1) / TB), TB, 0, stream>>>(np, neidx, nea);
}
